// TejaGPT_42795054137839
// MI455X (gfx1250) — compile-verified, matrix operand order checked by reading, not run on hardware
//
#include <hip/hip_runtime.h>
#include <hip/hip_bf16.h>
#include <math.h>

// ---------------- CDNA5 WMMA types ----------------
typedef __attribute__((ext_vector_type(16))) __bf16 v16bf;
typedef __attribute__((ext_vector_type(8)))  __bf16 v8bf;
typedef __attribute__((ext_vector_type(8)))  float  v8f;

#define D_MODEL 768
#define N_HEAD  12
#define HEAD_DIM 64
#define SEQ_T   1024
#define N_B     2
#define ROWS    (N_B * SEQ_T)   // 2048
#define FF_DIM  3072
#define N_LAYER 12

// A/B 16-bit fragment k index for lane-half (ISA 7.12.2, 16-bit A 16x32):
// element i -> k = (i<8 ? half*8+i : 16+half*8+(i-8))
__device__ __forceinline__ int kmap(int i, int half) {
  return (i < 8) ? (half * 8 + i) : (16 + half * 8 + (i - 8));
}

// Load one 16-element bf16 fragment (contiguous [K] layout) with 2 x b128.
__device__ __forceinline__ v16bf load_frag(const __bf16* p, int half) {
  v8bf lo = *(const v8bf*)(p + half * 8);
  v8bf hi = *(const v8bf*)(p + 16 + half * 8);
  return __builtin_shufflevector(lo, hi, 0, 1, 2, 3, 4, 5, 6, 7, 8, 9, 10, 11, 12, 13,
                                 14, 15);
}

__device__ __forceinline__ v8f wmma_bf16(v16bf a, v16bf b, v8f c) {
  return __builtin_amdgcn_wmma_f32_16x16x32_bf16(false, a, false, b, (short)0, c,
                                                 false, false);
}

// ---------------- embedding: x = W_emb[idx] + W_pos[t] (f32) ----------------
__global__ __launch_bounds__(256) void embed_kernel(const int* __restrict__ idx,
                                                    const float* __restrict__ Wemb,
                                                    const float* __restrict__ Wpos,
                                                    float* __restrict__ x) {
  const int bt = blockIdx.x;
  const int t  = bt % SEQ_T;
  const int tok = idx[bt];
  const float* e = Wemb + (size_t)tok * D_MODEL;
  const float* p = Wpos + (size_t)t * D_MODEL;
  float* xo = x + (size_t)bt * D_MODEL;
  for (int i = threadIdx.x; i < D_MODEL; i += 256) xo[i] = e[i] + p[i];
}

// ---------------- f32 -> bf16 flat convert (W_emb) ----------------
__global__ __launch_bounds__(256) void cvt_bf16_kernel(const float* __restrict__ in,
                                                       __bf16* __restrict__ out,
                                                       int total) {
  int i = (blockIdx.x * 256 + threadIdx.x) * 4;
#pragma unroll
  for (int j = 0; j < 4; ++j)
    if (i + j < total) out[i + j] = (__bf16)in[i + j];
}

// ------------- f32 [K,N] -> bf16 [N,K] tiled transpose-convert -------------
__global__ __launch_bounds__(256) void transpose_bf16_kernel(const float* __restrict__ in,
                                                             __bf16* __restrict__ out,
                                                             int K, int N) {
  __shared__ __bf16 tile[32][33];
  const int kt = blockIdx.y * 32, nt = blockIdx.x * 32;
  for (int r = threadIdx.y; r < 32; r += 8)
    tile[r][threadIdx.x] = (__bf16)in[(size_t)(kt + r) * N + nt + threadIdx.x];
  __syncthreads();
  for (int r = threadIdx.y; r < 32; r += 8)
    out[(size_t)(nt + r) * K + kt + threadIdx.x] = tile[threadIdx.x][r];
}

// ---------------- layernorm: f32 in -> bf16 out, block per row --------------
__global__ __launch_bounds__(256) void ln_kernel(const float* __restrict__ x,
                                                 const float* __restrict__ g,
                                                 const float* __restrict__ bt,
                                                 __bf16* __restrict__ out) {
  __shared__ float red[256];
  const int row = blockIdx.x;
  const float* xr = x + (size_t)row * D_MODEL;

  float s = 0.f;
  for (int i = threadIdx.x; i < D_MODEL; i += 256) s += xr[i];
  red[threadIdx.x] = s; __syncthreads();
  for (int off = 128; off; off >>= 1) {
    if (threadIdx.x < off) red[threadIdx.x] += red[threadIdx.x + off];
    __syncthreads();
  }
  const float mu = red[0] * (1.f / D_MODEL);
  __syncthreads();

  float v = 0.f;
  for (int i = threadIdx.x; i < D_MODEL; i += 256) { float d = xr[i] - mu; v += d * d; }
  red[threadIdx.x] = v; __syncthreads();
  for (int off = 128; off; off >>= 1) {
    if (threadIdx.x < off) red[threadIdx.x] += red[threadIdx.x + off];
    __syncthreads();
  }
  const float rstd = rsqrtf(red[0] * (1.f / D_MODEL) + 1e-5f);

  __bf16* orow = out + (size_t)row * D_MODEL;
  for (int i = threadIdx.x; i < D_MODEL; i += 256)
    orow[i] = (__bf16)((xr[i] - mu) * rstd * g[i] + bt[i]);
}

// ---------------- WMMA GEMM: all-bf16 operands, B stored [N,K] --------------
// Block = 256 threads = 8 waves, tile 128 rows x 64 cols. B k-chunk (64x32 bf16)
// is cooperatively staged in a double-buffered LDS panel shared by all 8 waves;
// fragments come from ds_load_b128. A rows stream from global (b128 x2/frag).
// OUTMODE 0: f32 [M,N]  1: bf16 [M,N]  2: bf16 per-head transposed [B][H][HD][T]
template <bool BIAS, bool APPLY_GELU, bool RESID, int OUTMODE>
__global__ __launch_bounds__(256) void gemm_bf16_k(const __bf16* __restrict__ A,
                                                   const __bf16* __restrict__ B,
                                                   const float* __restrict__ bias,
                                                   const float* __restrict__ resid,
                                                   void* __restrict__ Cout,
                                                   int M, int N, int K) {
  __shared__ __bf16 bs[2][64][40];  // row pad 40 (80B) keeps 16B-aligned b128 ops
  const int tid  = threadIdx.x;
  const int lane = tid & 31;
  const int wave = tid >> 5;
  const int half = lane >> 4;
  const int ln   = lane & 15;
  const int m0 = blockIdx.y * 128 + wave * 16;
  const int n0 = blockIdx.x * 64;

  // cooperative B staging map: thread -> (column 0..63, 8-elem part 0..3)
  const int sc = tid >> 2;
  const int sp = tid & 3;
  int bcol = n0 + sc;
  if (bcol >= N) bcol = N - 1;  // clamp (logits tail); results discarded at store
  const __bf16* Bs = B + (size_t)bcol * K + sp * 8;

  v8f acc[4];
#pragma unroll
  for (int t = 0; t < 4; ++t) acc[t] = (v8f){0.f, 0.f, 0.f, 0.f, 0.f, 0.f, 0.f, 0.f};

  const __bf16* Ap = A + (size_t)(m0 + ln) * K;

  // preload k-chunk 0
  *(v8bf*)(&bs[0][sc][sp * 8]) = *(const v8bf*)(Bs);
  int p = 0;
  for (int k0 = 0; k0 < K; k0 += 32) {
    __syncthreads();  // bs[p] for chunk k0 is visible block-wide
    if (k0 + 32 < K) {
      *(v8bf*)(&bs[p ^ 1][sc][sp * 8]) = *(const v8bf*)(Bs + k0 + 32);
      __builtin_prefetch(Bs + k0 + 64, 0, 0);  // global_prefetch_b8
    }
    const v16bf a = load_frag(Ap + k0, half);
    __builtin_prefetch(Ap + k0 + 32, 0, 0);
#pragma unroll
    for (int t = 0; t < 4; ++t) {
      const v16bf b = load_frag(&bs[p][t * 16 + ln][0], half);  // 2 x ds_load_b128
      acc[t] = wmma_bf16(a, b, acc[t]);
    }
    p ^= 1;
  }

#pragma unroll
  for (int t = 0; t < 4; ++t) {
    const int col = n0 + t * 16 + ln;
    if (col < N) {
      const float bv = BIAS ? bias[col] : 0.f;
#pragma unroll
      for (int r = 0; r < 8; ++r) {
        const int row = m0 + r + 8 * half;
        float val = acc[t][r] + bv;
        if (APPLY_GELU) val = 0.5f * val * (1.f + erff(val * 0.70710678118654752f));
        if (RESID) val += resid[(size_t)row * N + col];
        if (OUTMODE == 0) {
          ((float*)Cout)[(size_t)row * N + col] = val;
        } else if (OUTMODE == 1) {
          ((__bf16*)Cout)[(size_t)row * N + col] = (__bf16)val;
        } else {  // V transposed per head: [B][H][HD][T]
          const int hh = col >> 6, dd = col & (HEAD_DIM - 1);
          const int bb = row >> 10, tt = row & (SEQ_T - 1);
          ((__bf16*)Cout)[(((size_t)bb * N_HEAD + hh) * HEAD_DIM + dd) * SEQ_T + tt] =
              (__bf16)val;
        }
      }
    }
  }
}

// -------- flash attention (causal): q,k bf16 [B,T,D], v bf16 [B,H,HD,T] ------
__global__ __launch_bounds__(128) void attn_kernel(const __bf16* __restrict__ q,
                                                   const __bf16* __restrict__ k,
                                                   const __bf16* __restrict__ vT,
                                                   __bf16* __restrict__ o) {
  const int b = blockIdx.z, h = blockIdx.y;
  const int wave = threadIdx.x >> 5;
  const int lane = threadIdx.x & 31;
  const int half = lane >> 4;
  const int ln   = lane & 15;
  const int m0   = blockIdx.x * 64 + wave * 16;
  const float scale = 0.036084391824351615f;  // 768^-0.5 (full n_embd, per ref)

  __shared__ __bf16 plds[4][16][32];

  const __bf16* qrow = q + ((size_t)(b * SEQ_T + m0 + ln)) * D_MODEL + h * HEAD_DIM;
  v16bf aq[2];
#pragma unroll
  for (int kk = 0; kk < 2; ++kk) aq[kk] = load_frag(qrow + kk * 32, half);

  float m_run[8], l_run[8];
  v8f oacc[4];
#pragma unroll
  for (int r = 0; r < 8; ++r) { m_run[r] = -3e38f; l_run[r] = 0.f; }
#pragma unroll
  for (int t = 0; t < 4; ++t) oacc[t] = (v8f){0.f, 0.f, 0.f, 0.f, 0.f, 0.f, 0.f, 0.f};

  for (int s0 = 0; s0 < m0 + 16; s0 += 32) {
    // ---- scores: two 16-key tiles -> 16x32 ----
    v8f sc[2];
#pragma unroll
    for (int ts = 0; ts < 2; ++ts) {
      v8f s = (v8f){0.f, 0.f, 0.f, 0.f, 0.f, 0.f, 0.f, 0.f};
      const int key = s0 + ts * 16 + ln;
      const __bf16* krow = k + ((size_t)(b * SEQ_T + key)) * D_MODEL + h * HEAD_DIM;
#pragma unroll
      for (int kk = 0; kk < 2; ++kk)
        s = wmma_bf16(aq[kk], load_frag(krow + kk * 32, half), s);
      sc[ts] = s;
    }
    // ---- scale + causal mask + running row-max ----
    float tmax[8];
#pragma unroll
    for (int r = 0; r < 8; ++r) {
      const int row = m0 + r + 8 * half;
#pragma unroll
      for (int ts = 0; ts < 2; ++ts) {
        const int col = s0 + ts * 16 + ln;
        const float sv = sc[ts][r] * scale;
        sc[ts][r] = (col <= row) ? sv : -3e38f;
      }
      tmax[r] = fmaxf(sc[0][r], sc[1][r]);
    }
#pragma unroll
    for (int off = 1; off < 16; off <<= 1)
#pragma unroll
      for (int r = 0; r < 8; ++r) tmax[r] = fmaxf(tmax[r], __shfl_xor(tmax[r], off, 32));

    float alpha[8], lsum[8];
#pragma unroll
    for (int r = 0; r < 8; ++r) {
      const float nm = fmaxf(m_run[r], tmax[r]);
      alpha[r] = __expf(m_run[r] - nm);
      m_run[r] = nm;
      const float p0 = __expf(sc[0][r] - nm);
      const float p1 = __expf(sc[1][r] - nm);
      sc[0][r] = p0; sc[1][r] = p1;
      lsum[r] = p0 + p1;
    }
#pragma unroll
    for (int off = 1; off < 16; off <<= 1)
#pragma unroll
      for (int r = 0; r < 8; ++r) lsum[r] += __shfl_xor(lsum[r], off, 32);
#pragma unroll
    for (int r = 0; r < 8; ++r) {
      l_run[r] = l_run[r] * alpha[r] + lsum[r];
#pragma unroll
      for (int t = 0; t < 4; ++t) oacc[t][r] *= alpha[r];
    }

    // ---- transpose P (C-layout) -> A-fragment layout via LDS ----
#pragma unroll
    for (int r = 0; r < 8; ++r) {
      const int rl = r + 8 * half;
      plds[wave][rl][ln]      = (__bf16)sc[0][r];
      plds[wave][rl][16 + ln] = (__bf16)sc[1][r];
    }
    __builtin_amdgcn_wave_barrier();
    asm volatile("s_wait_dscnt 0" ::: "memory");  // per-wave LDS RAW fence
    const v16bf ap = load_frag(&plds[wave][ln][0], half);  // 2 x ds_load_b128
    __builtin_amdgcn_wave_barrier();

    // ---- O += P @ V : vT rows contiguous along key index ----
#pragma unroll
    for (int t = 0; t < 4; ++t) {
      const __bf16* vrow =
          vT + (((size_t)b * N_HEAD + h) * HEAD_DIM + t * 16 + ln) * SEQ_T + s0;
      oacc[t] = wmma_bf16(ap, load_frag(vrow, half), oacc[t]);
    }
  }

  // ---- normalize + store (bf16 row-major [B,T,D]) ----
#pragma unroll
  for (int r = 0; r < 8; ++r) {
    const int row = m0 + r + 8 * half;
    const float inv = 1.f / l_run[r];
#pragma unroll
    for (int t = 0; t < 4; ++t)
      o[((size_t)(b * SEQ_T + row)) * D_MODEL + h * HEAD_DIM + t * 16 + ln] =
          (__bf16)(oacc[t][r] * inv);
  }
}

// ---------------- host orchestration ----------------
extern "C" void kernel_launch(void* const* d_in, const int* in_sizes, int n_in,
                              void* d_out, int out_size, void* d_ws, size_t ws_size,
                              hipStream_t stream) {
  (void)n_in; (void)out_size; (void)ws_size;
  const int*   idx   = (const int*)d_in[0];
  const float* W_emb = (const float*)d_in[1];
  const float* W_pos = (const float*)d_in[2];
  const float* Wq    = (const float*)d_in[3];
  const float* Wk    = (const float*)d_in[4];
  const float* Wv    = (const float*)d_in[5];
  const float* Wo    = (const float*)d_in[6];
  const float* bo    = (const float*)d_in[7];
  const float* W1    = (const float*)d_in[8];
  const float* b1    = (const float*)d_in[9];
  const float* W2    = (const float*)d_in[10];
  const float* b2    = (const float*)d_in[11];
  const float* ln1_g = (const float*)d_in[12];
  const float* ln1_b = (const float*)d_in[13];
  const float* ln2_g = (const float*)d_in[14];
  const float* ln2_b = (const float*)d_in[15];
  const float* lnf_g = (const float*)d_in[16];
  const float* lnf_b = (const float*)d_in[17];
  const int V = in_sizes[1] / D_MODEL;  // 50261

  // ---- workspace carve-up (256B aligned slices) ----
  char* wp = (char*)d_ws;
  auto carve = [&wp](size_t bytes) -> void* {
    void* p = (void*)wp;
    wp += (bytes + 255) & ~(size_t)255;
    return p;
  };
  const size_t NX = (size_t)ROWS * D_MODEL;
  float*  x     = (float*)carve(NX * 4);
  float*  xs    = (float*)carve(NX * 4);
  __bf16* hbuf  = (__bf16*)carve(NX * 2);
  __bf16* qb    = (__bf16*)carve(NX * 2);
  __bf16* kb    = (__bf16*)carve(NX * 2);
  __bf16* vTb   = (__bf16*)carve(NX * 2);
  __bf16* ob    = (__bf16*)carve(NX * 2);
  __bf16* ff1b  = (__bf16*)carve((size_t)ROWS * FF_DIM * 2);
  __bf16* wembB = (__bf16*)carve((size_t)V * D_MODEL * 2);
  __bf16* wqT   = (__bf16*)carve((size_t)D_MODEL * D_MODEL * 2);
  __bf16* wkT   = (__bf16*)carve((size_t)D_MODEL * D_MODEL * 2);
  __bf16* wvT   = (__bf16*)carve((size_t)D_MODEL * D_MODEL * 2);
  __bf16* woT   = (__bf16*)carve((size_t)D_MODEL * D_MODEL * 2);
  __bf16* w1T   = (__bf16*)carve((size_t)D_MODEL * FF_DIM * 2);
  __bf16* w2T   = (__bf16*)carve((size_t)D_MODEL * FF_DIM * 2);

  const dim3 tb(32, 8);
  const dim3 gdd(D_MODEL / 32, D_MODEL / 32);   // [768,768] transpose
  const dim3 gdf(FF_DIM / 32, D_MODEL / 32);    // W1 [768,3072] -> [3072,768]
  const dim3 gfd(D_MODEL / 32, FF_DIM / 32);    // W2 [3072,768] -> [768,3072]

  const dim3 g768((D_MODEL + 63) / 64, ROWS / 128);
  const dim3 gff((FF_DIM + 63) / 64, ROWS / 128);
  const dim3 gattn(SEQ_T / 64, N_HEAD, N_B);

  embed_kernel<<<ROWS, 256, 0, stream>>>(idx, W_emb, W_pos, x);
  {
    const int total = V * D_MODEL;
    cvt_bf16_kernel<<<(total / 4 + 255) / 256, 256, 0, stream>>>(W_emb, wembB, total);
  }

  for (int l = 0; l < N_LAYER; ++l) {
    const size_t wdd = (size_t)l * D_MODEL * D_MODEL;
    const size_t wdf = (size_t)l * D_MODEL * FF_DIM;

    // stage this layer's weights as bf16 [N,K]
    transpose_bf16_kernel<<<gdd, tb, 0, stream>>>(Wq + wdd, wqT, D_MODEL, D_MODEL);
    transpose_bf16_kernel<<<gdd, tb, 0, stream>>>(Wk + wdd, wkT, D_MODEL, D_MODEL);
    transpose_bf16_kernel<<<gdd, tb, 0, stream>>>(Wv + wdd, wvT, D_MODEL, D_MODEL);
    transpose_bf16_kernel<<<gdd, tb, 0, stream>>>(Wo + wdd, woT, D_MODEL, D_MODEL);
    transpose_bf16_kernel<<<gdf, tb, 0, stream>>>(W1 + wdf, w1T, D_MODEL, FF_DIM);
    transpose_bf16_kernel<<<gfd, tb, 0, stream>>>(W2 + wdf, w2T, FF_DIM, D_MODEL);

    ln_kernel<<<ROWS, 256, 0, stream>>>(x, ln1_g + l * D_MODEL, ln1_b + l * D_MODEL, hbuf);
    gemm_bf16_k<false, false, false, 1><<<g768, 256, 0, stream>>>(
        hbuf, wqT, nullptr, nullptr, qb, ROWS, D_MODEL, D_MODEL);
    gemm_bf16_k<false, false, false, 1><<<g768, 256, 0, stream>>>(
        hbuf, wkT, nullptr, nullptr, kb, ROWS, D_MODEL, D_MODEL);
    gemm_bf16_k<false, false, false, 2><<<g768, 256, 0, stream>>>(
        hbuf, wvT, nullptr, nullptr, vTb, ROWS, D_MODEL, D_MODEL);

    attn_kernel<<<gattn, 128, 0, stream>>>(qb, kb, vTb, ob);

    // xs = x + (o @ Wo + bo)         (f32 out)
    gemm_bf16_k<true, false, true, 0><<<g768, 256, 0, stream>>>(
        ob, woT, bo + (size_t)l * D_MODEL, x, xs, ROWS, D_MODEL, D_MODEL);

    ln_kernel<<<ROWS, 256, 0, stream>>>(xs, ln2_g + l * D_MODEL, ln2_b + l * D_MODEL, hbuf);
    // ff1 = gelu(h2 @ W1 + b1)       (bf16 out)
    gemm_bf16_k<true, true, false, 1><<<gff, 256, 0, stream>>>(
        hbuf, w1T, b1 + (size_t)l * FF_DIM, nullptr, ff1b, ROWS, FF_DIM, D_MODEL);
    // x = x + (ff1 @ W2 + b2)        (residual is x, NOT xs — faithful to ref)
    gemm_bf16_k<true, false, true, 0><<<g768, 256, 0, stream>>>(
        ff1b, w2T, b2 + (size_t)l * D_MODEL, x, x, ROWS, D_MODEL, FF_DIM);
  }

  ln_kernel<<<ROWS, 256, 0, stream>>>(x, lnf_g, lnf_b, hbuf);
  const dim3 glog((V + 63) / 64, ROWS / 128);
  gemm_bf16_k<false, false, false, 0><<<glog, 256, 0, stream>>>(
      hbuf, wembB, nullptr, nullptr, (float*)d_out, ROWS, V, D_MODEL);
}
